// RelativePositionalAttention_18098992185512
// MI455X (gfx1250) — matrix operand order repaired and checked by reading, not compiled, not run on hardware
//
#include <hip/hip_runtime.h>
#include <cstdint>
#include <cstddef>

typedef __attribute__((ext_vector_type(16))) __bf16 v16bf;
typedef __attribute__((ext_vector_type(8)))  float  v8f;

namespace {
constexpr int kB = 4, kN = 1024, kD = 512, kH = 8, kHD = 64;
constexpr int kRows = kB * kN;               // 4096
constexpr int kXElems = kRows * kD;          // 2,097,152 per activation matrix
constexpr int kWElems = kD * kD;             // 262,144 per weight matrix
constexpr int kXChunks = kXElems / 8;        // 262,144
constexpr int kWChunks = kWElems / 8;        // 32,768
constexpr float kScale = 0.125f;             // 1/sqrt(HD)
constexpr float kInvLogStep = 5.58651987f;   // 31 / ln(257)  (exp-spaced spatial bins)

union BF16Frag {
  uint4 q[2];
  unsigned short s[16];
  v16bf v;
};

__device__ __forceinline__ unsigned short f2bf(float f) {
  unsigned u = __builtin_bit_cast(unsigned, f);
  unsigned r = (u + 0x7FFFu + ((u >> 16) & 1u)) >> 16;  // round-to-nearest-even
  return (unsigned short)r;
}

__device__ __forceinline__ v8f wmma_bf16(const BF16Frag& a, const BF16Frag& b, v8f c) {
  // (neg_a, A, neg_b, B, c_mod, C, reuse_a, reuse_b)
  return __builtin_amdgcn_wmma_f32_16x16x32_bf16(false, a.v, false, b.v, (short)0, c,
                                                 false, false);
}
} // namespace

// ---------------------------------------------------------------------------
// Kernel 0: one-shot f32 -> bf16 conversion of activations and weights.
// Memory-bound streaming pass; keeps ALL conversion VALU out of GEMM loops.
// Each thread: 8 elements (2x float4 in, 1x uint4 out).
// ---------------------------------------------------------------------------
__global__ __launch_bounds__(256) void cvt_kernel(
    const float* __restrict__ q, const float* __restrict__ k,
    const float* __restrict__ v,
    const float* __restrict__ Wq, const float* __restrict__ Wk,
    const float* __restrict__ Wv, const float* __restrict__ Wo,
    unsigned short* __restrict__ xq, unsigned short* __restrict__ xk,
    unsigned short* __restrict__ xv,
    unsigned short* __restrict__ wq, unsigned short* __restrict__ wk,
    unsigned short* __restrict__ wv, unsigned short* __restrict__ wo) {
  const int idx = blockIdx.x * 256 + threadIdx.x;
  const float* src;
  unsigned short* dst;
  size_t off;
  if (idx < 3 * kXChunks) {
    const int m = idx / kXChunks;
    off = (size_t)(idx - m * kXChunks) * 8;
    src = (m == 0) ? q : (m == 1) ? k : v;
    dst = (m == 0) ? xq : (m == 1) ? xk : xv;
  } else {
    const int widx = idx - 3 * kXChunks;
    const int m = widx / kWChunks;
    off = (size_t)(widx - m * kWChunks) * 8;
    src = (m == 0) ? Wq : (m == 1) ? Wk : (m == 2) ? Wv : Wo;
    dst = (m == 0) ? wq : (m == 1) ? wk : (m == 2) ? wv : wo;
  }
  const float4 a = *(const float4*)(src + off);
  const float4 b = *(const float4*)(src + off + 4);
  union { uint4 u; unsigned short s[8]; } pk;
  pk.s[0] = f2bf(a.x); pk.s[1] = f2bf(a.y); pk.s[2] = f2bf(a.z); pk.s[3] = f2bf(a.w);
  pk.s[4] = f2bf(b.x); pk.s[5] = f2bf(b.y); pk.s[6] = f2bf(b.z); pk.s[7] = f2bf(b.w);
  *(uint4*)(dst + off) = pk.u;
}

// ---------------------------------------------------------------------------
// Kernel 1: fused Q/K/V projection.  One wave -> 16 rows x 64 cols (one head).
// Ping-pong double-buffered bf16 b128 loads + WMMA (loads for step k+1 issue
// under the WMMAs of step k).  V stored transposed [b][h][hd][n].
// ---------------------------------------------------------------------------
__global__ __launch_bounds__(256) void qkv_proj_kernel(
    const unsigned short* __restrict__ xq, const unsigned short* __restrict__ xk,
    const unsigned short* __restrict__ xv,
    const unsigned short* __restrict__ wq, const unsigned short* __restrict__ wk,
    const unsigned short* __restrict__ wv,
    const float* __restrict__ bq, const float* __restrict__ bk,
    const float* __restrict__ bv,
    unsigned short* __restrict__ q_bf, unsigned short* __restrict__ k_bf,
    unsigned short* __restrict__ vT_bf) {
  const int wid  = (blockIdx.x * blockDim.x + threadIdx.x) >> 5;
  const int lane = threadIdx.x & 31;
  const int hf = lane >> 4, ln = lane & 15;

  const int mat = wid >> 11;           // 0=Q 1=K 2=V  (2048 wave-jobs each)
  const int rt  = (wid & 2047) >> 3;   // row tile [0,256)
  const int ct  = wid & 7;             // head / col tile [0,8)
  const int rowbase = rt * 16;
  const int colbase = ct * 64;

  const unsigned short* X = (mat == 0) ? xq : (mat == 1) ? xk : xv;
  const unsigned short* W = (mat == 0) ? wq : (mat == 1) ? wk : wv;
  const float* bias       = (mat == 0) ? bq : (mat == 1) ? bk : bv;

  v8f acc[4];
#pragma unroll
  for (int t = 0; t < 4; ++t) {
    const float bvv = bias[colbase + t * 16 + ln];
#pragma unroll
    for (int r = 0; r < 8; ++r) acc[t][r] = bvv;
  }

  const unsigned short* arow = X + (size_t)(rowbase + ln) * kD;
  const unsigned short* wrow = W + (size_t)(colbase + ln) * kD;

  BF16Frag a[2];        // A: K = hf*8 + e  and  +16
  BF16Frag wb[2][4];    // B: col = ln, K = hf*8 + e  and  +16
  auto load_step = [&](int ks, int buf) {
    a[buf].q[0] = *(const uint4*)(arow + ks + hf * 8);
    a[buf].q[1] = *(const uint4*)(arow + ks + hf * 8 + 16);
#pragma unroll
    for (int t = 0; t < 4; ++t) {
      const unsigned short* wp = wrow + (size_t)t * 16 * kD + ks + hf * 8;
      wb[buf][t].q[0] = *(const uint4*)(wp);
      wb[buf][t].q[1] = *(const uint4*)(wp + 16);
    }
  };

  load_step(0, 0);
#pragma unroll
  for (int ks = 0; ks < kD; ks += 32) {
    const int buf = (ks >> 5) & 1;
    if (ks + 32 < kD) load_step(ks + 32, buf ^ 1);   // prefetch next step
#pragma unroll
    for (int t = 0; t < 4; ++t) acc[t] = wmma_bf16(a[buf], wb[buf][t], acc[t]);
  }

  if (mat < 2) {                              // Q/K: row-major [b][h][n][hd]
    unsigned short* O = (mat == 0) ? q_bf : k_bf;
#pragma unroll
    for (int t = 0; t < 4; ++t) {
      const int col = colbase + t * 16 + ln;
      const int h = col >> 6, hd = col & 63;
#pragma unroll
      for (int r = 0; r < 8; ++r) {
        const int row = rowbase + hf * 8 + r;
        const int b = row >> 10, n = row & 1023;
        O[((size_t)(b * kH + h) * kN + n) * kHD + hd] = f2bf(acc[t][r]);
      }
    }
  } else {                                    // V: transposed [b][h][hd][n], packed b128
#pragma unroll
    for (int t = 0; t < 4; ++t) {
      const int col = colbase + t * 16 + ln;
      const int h = col >> 6, hd = col & 63;
      union { uint4 u; unsigned short s[8]; } pk;
#pragma unroll
      for (int r = 0; r < 8; ++r) pk.s[r] = f2bf(acc[t][r]);
      const int row0 = rowbase + hf * 8;
      const int b = row0 >> 10, n0 = row0 & 1023;
      *(uint4*)&vT_bf[((size_t)(b * kH + h) * kHD + hd) * kN + n0] = pk.u;
    }
  }
}

// ---------------------------------------------------------------------------
// Kernel 2: flash attention with analytic relative-position bias.
// One wave -> (b, h, 16 query rows), streaming 32-key blocks.
// ---------------------------------------------------------------------------
__global__ __launch_bounds__(256) void attn_kernel(
    const unsigned short* __restrict__ q_bf, const unsigned short* __restrict__ k_bf,
    const unsigned short* __restrict__ vT_bf, const float* __restrict__ coords,
    const float* __restrict__ bias_table, unsigned short* __restrict__ y_bf) {
  __shared__ unsigned short plds[8][16 * 32];   // per-wave P staging (C-layout -> A-layout)
  const int wid   = (blockIdx.x * blockDim.x + threadIdx.x) >> 5;
  const int wslot = threadIdx.x >> 5;
  const int lane  = threadIdx.x & 31;
  const int hf = lane >> 4, ln = lane & 15;

  const int b  = wid >> 9;         // 512 jobs per batch
  const int h  = (wid >> 6) & 7;
  const int qt = wid & 63;
  const int bh = b * kH + h;

  // Q A-fragments (HD=64 -> two K=32 fragments)
  BF16Frag qa[2];
  {
    const unsigned short* qp = q_bf + ((size_t)bh * kN + qt * 16 + ln) * kHD;
#pragma unroll
    for (int kst = 0; kst < 2; ++kst) {
      qa[kst].q[0] = *(const uint4*)(qp + kst * 32 + hf * 8);
      qa[kst].q[1] = *(const uint4*)(qp + kst * 32 + hf * 8 + 16);
    }
  }

  // query coords for this lane's 8 C-layout rows (M = hf*8 + r)
  float qtv[8], qyv[8], qxv[8];
#pragma unroll
  for (int r = 0; r < 8; ++r) {
    const int row = qt * 16 + hf * 8 + r;
    const float* cp = coords + ((size_t)b * kN + row) * 3;
    qtv[r] = cp[0]; qyv[r] = cp[1]; qxv[r] = cp[2];
  }

  float m[8], l[8];
#pragma unroll
  for (int r = 0; r < 8; ++r) { m[r] = -3.0e38f; l[r] = 0.0f; }
  v8f acc[4];
#pragma unroll
  for (int t = 0; t < 4; ++t)
#pragma unroll
    for (int r = 0; r < 8; ++r) acc[t][r] = 0.0f;

  for (int jb = 0; jb < kN / 32; ++jb) {
    if (jb + 1 < kN / 32) {  // pull next key/value block toward the caches
      __builtin_prefetch(k_bf + ((size_t)bh * kN + (jb + 1) * 32 + ln) * kHD, 0, 1);
      __builtin_prefetch(vT_bf + ((size_t)bh * kHD + ln) * kN + (jb + 1) * 32, 0, 1);
    }

    float s[2][8];
#pragma unroll
    for (int jt = 0; jt < 2; ++jt) {
      const int colbase = jb * 32 + jt * 16;
      const unsigned short* kp = k_bf + ((size_t)bh * kN + colbase + ln) * kHD;
      v8f c;
#pragma unroll
      for (int r = 0; r < 8; ++r) c[r] = 0.0f;
#pragma unroll
      for (int kst = 0; kst < 2; ++kst) {
        BF16Frag kb;                       // B: col = key ln, K = hd (hf*8 + e, +16)
        kb.q[0] = *(const uint4*)(kp + kst * 32 + hf * 8);
        kb.q[1] = *(const uint4*)(kp + kst * 32 + hf * 8 + 16);
        c = wmma_bf16(qa[kst], kb, c);
      }
      // bias/mask for this lane's key column
      const float* cp = coords + ((size_t)b * kN + colbase + ln) * 3;
      const float kt = cp[0], ky = cp[1], kx = cp[2];
#pragma unroll
      for (int r = 0; r < 8; ++r) {
        const float dt = qtv[r] - kt, dy = qyv[r] - ky, dx = qxv[r] - kx;
        const float sd2 = dy * dy + dx * dx;
        const float sd = sqrtf(sd2);
        const float fd = sqrtf(sd2 + dt * dt);
        int sidx = 0;                      // searchsorted(exp bins) == ceil(ln d / step)
        if (sd > 1.0f) sidx = min(31, (int)ceilf(__logf(sd) * kInvLogStep));
        const int tidx = max(0, min(32, (int)ceilf(dt + 16.0f)));
        const float bias = __ldg(bias_table + (sidx + tidx * 32) * kH + h);
        const float mask = (sd > 256.0f) ? -1000.0f : 0.0f;
        s[jt][r] = c[r] * kScale + mask + bias + __expf(-0.1f * fd);
      }
    }

    // online softmax (rows live in one 16-lane half -> width-16 xor reductions)
#pragma unroll
    for (int r = 0; r < 8; ++r) {
      float mx = fmaxf(s[0][r], s[1][r]);
#pragma unroll
      for (int d = 1; d <= 8; d <<= 1) mx = fmaxf(mx, __shfl_xor(mx, d, 16));
      const float mn = fmaxf(m[r], mx);
      const float p0 = __expf(s[0][r] - mn);
      const float p1 = __expf(s[1][r] - mn);
      float rs = p0 + p1;
#pragma unroll
      for (int d = 1; d <= 8; d <<= 1) rs += __shfl_xor(rs, d, 16);
      const float alpha = __expf(m[r] - mn);
      l[r] = l[r] * alpha + rs;
      m[r] = mn;
      const int prow = hf * 8 + r;
      plds[wslot][prow * 32 + ln]      = f2bf(p0);
      plds[wslot][prow * 32 + 16 + ln] = f2bf(p1);
#pragma unroll
      for (int t = 0; t < 4; ++t) acc[t][r] *= alpha;
    }

    // reload P in WMMA A-matrix layout (same-wave DS ops are in order)
    BF16Frag pa;
    pa.q[0] = *(const uint4*)&plds[wslot][ln * 32 + hf * 8];
    pa.q[1] = *(const uint4*)&plds[wslot][ln * 32 + 16 + hf * 8];

    // P (16x32) @ V (32x64): V-transposed gives two contiguous 16B B-operand loads (keys hf*8+e, +16)
#pragma unroll
    for (int t = 0; t < 4; ++t) {
      const unsigned short* vp =
          vT_bf + ((size_t)bh * kHD + t * 16 + ln) * kN + jb * 32 + hf * 8;
      BF16Frag vb;
      vb.q[0] = *(const uint4*)(vp);
      vb.q[1] = *(const uint4*)(vp + 16);
      acc[t] = wmma_bf16(pa, vb, acc[t]);
    }
  }

  // normalize and store y as bf16, row-major [b][n][D]
#pragma unroll
  for (int r = 0; r < 8; ++r) {
    const float inv = 1.0f / l[r];
    const int row = qt * 16 + hf * 8 + r;
#pragma unroll
    for (int t = 0; t < 4; ++t) {
      y_bf[((size_t)b * kN + row) * kD + h * 64 + t * 16 + ln] = f2bf(acc[t][r] * inv);
    }
  }
}

// ---------------------------------------------------------------------------
// Kernel 3: output projection  out = y @ Wo^T + bo  (f32 result)
// ---------------------------------------------------------------------------
__global__ __launch_bounds__(256) void out_proj_kernel(
    const unsigned short* __restrict__ y_bf, const unsigned short* __restrict__ wo,
    const float* __restrict__ bo, float* __restrict__ out) {
  const int wid  = (blockIdx.x * blockDim.x + threadIdx.x) >> 5;
  const int lane = threadIdx.x & 31;
  const int hf = lane >> 4, ln = lane & 15;
  const int rt = wid >> 3;       // [0,256)
  const int ct = wid & 7;        // [0,8)
  const int rowbase = rt * 16, colbase = ct * 64;

  v8f acc[4];
#pragma unroll
  for (int t = 0; t < 4; ++t) {
    const float bvv = bo[colbase + t * 16 + ln];
#pragma unroll
    for (int r = 0; r < 8; ++r) acc[t][r] = bvv;
  }

  const unsigned short* arow = y_bf + (size_t)(rowbase + ln) * kD;
  const unsigned short* wrow = wo + (size_t)(colbase + ln) * kD;

  BF16Frag a[2];
  BF16Frag wb[2][4];
  auto load_step = [&](int ks, int buf) {
    a[buf].q[0] = *(const uint4*)(arow + ks + hf * 8);
    a[buf].q[1] = *(const uint4*)(arow + ks + hf * 8 + 16);
#pragma unroll
    for (int t = 0; t < 4; ++t) {
      const unsigned short* wp = wrow + (size_t)t * 16 * kD + ks + hf * 8;
      wb[buf][t].q[0] = *(const uint4*)(wp);
      wb[buf][t].q[1] = *(const uint4*)(wp + 16);
    }
  };

  load_step(0, 0);
#pragma unroll
  for (int ks = 0; ks < kD; ks += 32) {
    const int buf = (ks >> 5) & 1;
    if (ks + 32 < kD) load_step(ks + 32, buf ^ 1);   // prefetch next step
#pragma unroll
    for (int t = 0; t < 4; ++t) acc[t] = wmma_bf16(a[buf], wb[buf][t], acc[t]);
  }

#pragma unroll
  for (int t = 0; t < 4; ++t) {
    const int col = colbase + t * 16 + ln;
#pragma unroll
    for (int r = 0; r < 8; ++r) {
      const int row = rowbase + hf * 8 + r;
      out[(size_t)row * kD + col] = acc[t][r];
    }
  }
}

// ---------------------------------------------------------------------------
extern "C" void kernel_launch(void* const* d_in, const int* in_sizes, int n_in,
                              void* d_out, int out_size, void* d_ws, size_t ws_size,
                              hipStream_t stream) {
  (void)in_sizes; (void)n_in; (void)out_size; (void)ws_size;
  const float* query  = (const float*)d_in[0];
  const float* key    = (const float*)d_in[1];
  const float* value  = (const float*)d_in[2];
  const float* coords = (const float*)d_in[3];
  const float* Wq = (const float*)d_in[4];   const float* bq = (const float*)d_in[5];
  const float* Wk = (const float*)d_in[6];   const float* bk = (const float*)d_in[7];
  const float* Wv = (const float*)d_in[8];   const float* bv = (const float*)d_in[9];
  const float* Wo = (const float*)d_in[10];  const float* bo = (const float*)d_in[11];
  const float* bias_table = (const float*)d_in[12];

  // workspace layout (bf16 elements):
  //   xq | xk | xv           : 3 * kXElems   (converted activations)
  //   wq | wk | wv | wo      : 4 * kWElems   (converted weights)
  //   q_bf | k_bf | vT | y   : 4 * kXElems   (projected tensors + attn output)
  unsigned short* p = (unsigned short*)d_ws;
  unsigned short* xq_bf = p;              p += kXElems;
  unsigned short* xk_bf = p;              p += kXElems;
  unsigned short* xv_bf = p;              p += kXElems;
  unsigned short* wq_bf = p;              p += kWElems;
  unsigned short* wk_bf = p;              p += kWElems;
  unsigned short* wv_bf = p;              p += kWElems;
  unsigned short* wo_bf = p;              p += kWElems;
  unsigned short* q_bf  = p;              p += kXElems;
  unsigned short* k_bf  = p;              p += kXElems;
  unsigned short* vT_bf = p;              p += kXElems;
  unsigned short* y_bf  = p;

  // (3*262144 + 4*32768) chunks / 256 threads = 3584 blocks
  cvt_kernel<<<3584, 256, 0, stream>>>(query, key, value, Wq, Wk, Wv, Wo,
                                       xq_bf, xk_bf, xv_bf,
                                       wq_bf, wk_bf, wv_bf, wo_bf);
  // 3*256*8 = 6144 wave-jobs, 8 waves per 256-thread block -> 768 blocks
  qkv_proj_kernel<<<768, 256, 0, stream>>>(xq_bf, xk_bf, xv_bf,
                                           wq_bf, wk_bf, wv_bf,
                                           bq, bk, bv, q_bf, k_bf, vT_bf);
  // 4*8*64 = 2048 wave-jobs -> 256 blocks
  attn_kernel<<<256, 256, 0, stream>>>(q_bf, k_bf, vT_bf, coords, bias_table, y_bf);
  // 256*8 = 2048 wave-jobs -> 256 blocks
  out_proj_kernel<<<256, 256, 0, stream>>>(y_bf, wo_bf, bo, (float*)d_out);
}
